// GraphMemoryVQ_77970836292120
// MI455X (gfx1250) — compile-verified, matrix operand order checked by reading, not run on hardware
//
#include <hip/hip_runtime.h>
#include <hip/hip_bf16.h>
#include <stdint.h>


#define N_TOKENS 32768
#define DIMC     256
#define FDIM     512
#define NSYM     4096
#define KT       (FDIM / 32)
#define KT8      (FDIM / 128)
#define STILES   (NSYM / 16)
#define MTILES   (N_TOKENS / 16)
#define ROWS_PER_BLOCK 32
#define MBLOCKS  (N_TOKENS / ROWS_PER_BLOCK)
#define ZC_FLOATS ((size_t)2 * N_TOKENS * DIMC)

typedef __attribute__((ext_vector_type(16))) __bf16 v16bf;
typedef __attribute__((ext_vector_type(16))) int    v16i;
typedef __attribute__((ext_vector_type(8)))  float  v8f;

#define WMMA_BF16(A, B, C) \
    __builtin_amdgcn_wmma_f32_16x16x32_bf16(false, (A), false, (B), (short)0, (C), false, false)
#define WMMA_FP8(A, B, C) \
    __builtin_amdgcn_wmma_f32_16x16x128_fp8_fp8((A), (B), (short)0, (C), false, false)

__device__ __forceinline__ int frag_k(int p, int j, int laneHalf8) {
    int kk = (p < 4) ? (2 * p) : (16 + 2 * (p - 4));
    return kk + j + laneHalf8;
}

__device__ __forceinline__ int frag_k8(int v, int j, int laneHalf8) {
    const int v7 = v & 7;
    int kk = ((v7 & 1) * 4) + (((v7 >> 1) & 1) * 16) + ((v7 >> 2) * 32);
    return kk + j + laneHalf8 + (v >> 3) * 64;
}

__device__ __forceinline__ unsigned f32_to_e4m3(float f) {
    const unsigned u = __float_as_uint(f);
    const unsigned sign = (u >> 31) << 7;
    float a = __builtin_fabsf(f);
    if (!(a > 0.f)) return sign;
    if (a >= 448.f) return sign | 0x7Eu;
    const int e = (int)((u >> 23) & 0xFF) - 127;
    const int lsb = (e < -6) ? -9 : (e - 3);
    const float s = __uint_as_float((unsigned)((127 + lsb) << 23));
    const float q = rintf(a / s) * s;
    if (q == 0.f) return sign;
    const unsigned qu = __float_as_uint(q);
    const int qe = (int)((qu >> 23) & 0xFF) - 127;
    unsigned bits;
    if (qe < -6) bits = (unsigned)(q * 512.f);
    else         bits = (unsigned)((qe + 7) << 3) | ((qu >> 20) & 7);
    if (bits > 0x7Eu) bits = 0x7Eu;
    return sign | bits;
}

__global__ __launch_bounds__(128)
void pack_codebook_kernel(const float* __restrict__ cb,
                          v16bf* __restrict__ Bh16,
                          v16i*  __restrict__ B8h,
                          v16i*  __restrict__ B8l) {
    const int st = blockIdx.x;
    for (int e = threadIdx.x; e < KT * 32; e += blockDim.x) {
        const int kt = e >> 5, lane = e & 31;
        const int n = lane & 15, lh = (lane >> 4) * 8;
        const int s = st * 16 + n;
        v16bf vh;
        #pragma unroll
        for (int p = 0; p < 8; ++p)
            #pragma unroll
            for (int j = 0; j < 2; ++j) {
                const int k = kt * 32 + frag_k(p, j, lh);
                vh[p * 2 + j] = (__bf16)cb[(size_t)s * FDIM + k];
            }
        Bh16[((size_t)st * KT + kt) * 32 + lane] = vh;
    }
    for (int e = threadIdx.x; e < KT8 * 32; e += blockDim.x) {
        const int kt = e >> 5, lane = e & 31;
        const int n = lane & 15, lh = (lane >> 4) * 8;
        const int s = st * 16 + n;
        v16i vh, vl;
        #pragma unroll
        for (int v = 0; v < 16; ++v) {
            unsigned wh = 0, wl = 0;
            #pragma unroll
            for (int j = 0; j < 4; ++j) {
                const int k = kt * 128 + frag_k8(v, j, lh);
                const float f = cb[(size_t)s * FDIM + k];
                const float hi = (float)(__bf16)f;
                wh |= f32_to_e4m3(f) << (8 * j);
                wl |= f32_to_e4m3((f - hi) * 512.f) << (8 * j);
            }
            vh[v] = (int)wh;
            vl[v] = (int)wl;
        }
        const size_t fi = ((size_t)st * KT8 + kt) * 32 + lane;
        B8h[fi] = vh;
        B8l[fi] = vl;
    }
}

__global__ __launch_bounds__(128)
void pack_z_kernel(const float* __restrict__ zr,
                   const float* __restrict__ zi,
                   v16bf* __restrict__ Az16,
                   v16i*  __restrict__ Az8h,
                   v16i*  __restrict__ Az8l) {
    const int mt = blockIdx.x;
    for (int e = threadIdx.x; e < KT * 32; e += blockDim.x) {
        const int kt = e >> 5, lane = e & 31;
        const int m = lane & 15, lh = (lane >> 4) * 8;
        const size_t rbase = (size_t)(mt * 16 + m) * DIMC;
        v16bf vh;
        #pragma unroll
        for (int p = 0; p < 8; ++p)
            #pragma unroll
            for (int j = 0; j < 2; ++j) {
                const int k = kt * 32 + frag_k(p, j, lh);
                const float f = (k < DIMC) ? zr[rbase + k] : zi[rbase + k - DIMC];
                vh[p * 2 + j] = (__bf16)f;
            }
        Az16[((size_t)mt * KT + kt) * 32 + lane] = vh;
    }
    for (int e = threadIdx.x; e < KT8 * 32; e += blockDim.x) {
        const int kt = e >> 5, lane = e & 31;
        const int m = lane & 15, lh = (lane >> 4) * 8;
        const size_t rbase = (size_t)(mt * 16 + m) * DIMC;
        v16i vh, vl;
        #pragma unroll
        for (int v = 0; v < 16; ++v) {
            unsigned wh = 0, wl = 0;
            #pragma unroll
            for (int j = 0; j < 4; ++j) {
                const int k = kt * 128 + frag_k8(v, j, lh);
                const float f = (k < DIMC) ? zr[rbase + k] : zi[rbase + k - DIMC];
                const float hi = (float)(__bf16)f;
                wh |= f32_to_e4m3(f) << (8 * j);
                wl |= f32_to_e4m3((f - hi) * 512.f) << (8 * j);
            }
            vh[v] = (int)wh;
            vl[v] = (int)wl;
        }
        const size_t fi = ((size_t)mt * KT8 + kt) * 32 + lane;
        Az8h[fi] = vh;
        Az8l[fi] = vl;
    }
}

__global__ __launch_bounds__(128)
void cb_norms_kernel(const float* __restrict__ cb,
                     float* __restrict__ cbn,
                     float* __restrict__ loss_slot) {
    const int wave = threadIdx.x >> 5, lane = threadIdx.x & 31;
    const int row = blockIdx.x * 4 + wave;
    float s = 0.f;
    for (int k = lane; k < FDIM; k += 32) {
        const float f = cb[(size_t)row * FDIM + k];
        s += f * f;
    }
    #pragma unroll
    for (int m = 16; m; m >>= 1) s += __shfl_xor(s, m, 32);
    if (lane == 0) cbn[row] = s;
    if (blockIdx.x == 0 && threadIdx.x == 0) *loss_slot = 0.f;
}

__global__ __launch_bounds__(128, 4)
void vq_main_kernel(const float* __restrict__ zr,
                    const float* __restrict__ zi,
                    const int*   __restrict__ prev,
                    const float* __restrict__ cb,
                    const float* __restrict__ adj,
                    const v16bf* __restrict__ Bh16,
                    const v16i*  __restrict__ B8h,
                    const v16i*  __restrict__ B8l,
                    const v16bf* __restrict__ Az16,
                    const v16i*  __restrict__ Az8h,
                    const v16i*  __restrict__ Az8l,
                    const float* __restrict__ cbn,
                    float* __restrict__ out) {
    __shared__ v16bf sA16[2 * KT * 32];
    __shared__ v16i  sA8h[2 * KT8 * 32];
    __shared__ v16i  sA8l[2 * KT8 * 32];
    __shared__ unsigned long long sKey[ROWS_PER_BLOCK];
    __shared__ size_t sAdjOff[ROWS_PER_BLOCK];
    __shared__ int sIdx[ROWS_PER_BLOCK];
    __shared__ float sL[4];

    const int tid  = threadIdx.x;
    const int wave = tid >> 5, lane = tid & 31;
    const int row0 = blockIdx.x * ROWS_PER_BLOCK;

    if (tid < ROWS_PER_BLOCK) {
        sKey[tid] = ~0ull;
        sAdjOff[tid] = (size_t)prev[row0 + tid] * NSYM;
    }

    {
        const size_t a16 = (size_t)blockIdx.x * (2 * KT * 32);
        for (int e = tid; e < 2 * KT * 32; e += 128) sA16[e] = Az16[a16 + e];
        const size_t a8 = (size_t)blockIdx.x * (2 * KT8 * 32);
        for (int e = tid; e < 2 * KT8 * 32; e += 128) {
            sA8h[e] = Az8h[a8 + e];
            sA8l[e] = Az8l[a8 + e];
        }
    }
    __syncthreads();

    float minv[2][8];
    int   mini[2][8];
    #pragma unroll
    for (int mi = 0; mi < 2; ++mi)
        #pragma unroll
        for (int v = 0; v < 8; ++v) { minv[mi][v] = 3.4e38f; mini[mi][v] = 0; }

    const int lh = lane >> 4;
    const int n  = lane & 15;

    for (int pi = wave; pi < STILES / 2; pi += 4) {
        const int st0 = 2 * pi, st1 = st0 + 1;
        v8f acc00 = {}, acc01 = {}, acc10 = {}, acc11 = {};
        v8f cx00 = {}, cx01 = {}, cx10 = {}, cx11 = {};

        {
            const size_t b0 = (size_t)st0 * KT * 32 + lane;
            const size_t b1 = (size_t)st1 * KT * 32 + lane;
            #pragma unroll 2
            for (int kt = 0; kt < KT; ++kt) {
                const v16bf a0 = sA16[kt * 32 + lane];
                const v16bf a1 = sA16[KT * 32 + kt * 32 + lane];
                const v16bf bh0 = Bh16[b0 + kt * 32];
                const v16bf bh1 = Bh16[b1 + kt * 32];
                acc00 = WMMA_BF16(a0, bh0, acc00);
                acc01 = WMMA_BF16(a0, bh1, acc01);
                acc10 = WMMA_BF16(a1, bh0, acc10);
                acc11 = WMMA_BF16(a1, bh1, acc11);
            }
        }
        {
            const size_t b0 = (size_t)st0 * KT8 * 32 + lane;
            const size_t b1 = (size_t)st1 * KT8 * 32 + lane;
            #pragma unroll 1
            for (int kt = 0; kt < KT8; ++kt) {
                const v16i a0h = sA8h[kt * 32 + lane];
                const v16i a0l = sA8l[kt * 32 + lane];
                const v16i a1h = sA8h[KT8 * 32 + kt * 32 + lane];
                const v16i a1l = sA8l[KT8 * 32 + kt * 32 + lane];
                {
                    const v16i bh = B8h[b0 + kt * 32];
                    const v16i bl = B8l[b0 + kt * 32];
                    cx00 = WMMA_FP8(a0h, bl, cx00);
                    cx00 = WMMA_FP8(a0l, bh, cx00);
                    cx10 = WMMA_FP8(a1h, bl, cx10);
                    cx10 = WMMA_FP8(a1l, bh, cx10);
                }
                {
                    const v16i bh = B8h[b1 + kt * 32];
                    const v16i bl = B8l[b1 + kt * 32];
                    cx01 = WMMA_FP8(a0h, bl, cx01);
                    cx01 = WMMA_FP8(a0l, bh, cx01);
                    cx11 = WMMA_FP8(a1h, bl, cx11);
                    cx11 = WMMA_FP8(a1l, bh, cx11);
                }
            }
        }

        const int s0 = st0 * 16 + n;
        const int s1 = s0 + 16;
        const float cn0 = cbn[s0];
        const float cn1 = cbn[s1];
        const float inv512 = 1.f / 512.f;
        #pragma unroll
        for (int v = 0; v < 8; ++v) {
            const int m0 = v + lh * 8;
            const int m1 = m0 + 16;
            {
                const float dot = acc00[v] + cx00[v] * inv512;
                const float a = adj[sAdjOff[m0] + s0];
                const float sc = cn0 - 2.f * dot - 0.1f / (1.f + __expf(-a));
                if (sc < minv[0][v]) { minv[0][v] = sc; mini[0][v] = s0; }
            }
            {
                const float dot = acc01[v] + cx01[v] * inv512;
                const float a = adj[sAdjOff[m0] + s1];
                const float sc = cn1 - 2.f * dot - 0.1f / (1.f + __expf(-a));
                if (sc < minv[0][v]) { minv[0][v] = sc; mini[0][v] = s1; }
            }
            {
                const float dot = acc10[v] + cx10[v] * inv512;
                const float a = adj[sAdjOff[m1] + s0];
                const float sc = cn0 - 2.f * dot - 0.1f / (1.f + __expf(-a));
                if (sc < minv[1][v]) { minv[1][v] = sc; mini[1][v] = s0; }
            }
            {
                const float dot = acc11[v] + cx11[v] * inv512;
                const float a = adj[sAdjOff[m1] + s1];
                const float sc = cn1 - 2.f * dot - 0.1f / (1.f + __expf(-a));
                if (sc < minv[1][v]) { minv[1][v] = sc; mini[1][v] = s1; }
            }
        }
    }

    #pragma unroll
    for (int mi = 0; mi < 2; ++mi)
        #pragma unroll
        for (int v = 0; v < 8; ++v) {
            unsigned u = __float_as_uint(minv[mi][v]);
            u = (u & 0x80000000u) ? ~u : (u | 0x80000000u);
            unsigned long long key =
                ((unsigned long long)u << 32) | (unsigned)mini[mi][v];
            #pragma unroll
            for (int md = 8; md; md >>= 1) {
                const unsigned long long o =
                    (unsigned long long)__shfl_xor((long long)key, md, 16);
                if (o < key) key = o;
            }
            if (n == 0) atomicMin(&sKey[mi * 16 + v + lh * 8], key);
        }
    __syncthreads();

    if (tid < ROWS_PER_BLOCK) {
        const int idx = (int)(sKey[tid] & 0xFFFFFFFFu);
        sIdx[tid] = idx;
        out[ZC_FLOATS + 1 + row0 + tid] = (float)idx;
    }
    __syncthreads();

    float lsum = 0.f;
    for (int e = tid; e < ROWS_PER_BLOCK * FDIM; e += 128) {
        const int m = e >> 9, j = e & 511;
        const int row = row0 + m;
        const float q = cb[(size_t)sIdx[m] * FDIM + j];
        const float f = (j < DIMC) ? zr[(size_t)row * DIMC + j]
                                   : zi[(size_t)row * DIMC + j - DIMC];
        const float d = q - f;
        lsum += d * d;
        const size_t oi = ((size_t)row * DIMC + (j & (DIMC - 1))) * 2 + (j >> 8);
        out[oi] = q;
    }
    #pragma unroll
    for (int md = 16; md; md >>= 1) lsum += __shfl_xor(lsum, md, 32);
    if (lane == 0) sL[wave] = lsum;
    __syncthreads();
    if (tid == 0) {
        const float t = sL[0] + sL[1] + sL[2] + sL[3];
        atomicAdd(&out[ZC_FLOATS],
                  t * (1.25f / ((float)N_TOKENS * (float)FDIM)));
    }
}

extern "C" void kernel_launch(void* const* d_in, const int* in_sizes, int n_in,
                              void* d_out, int out_size, void* d_ws, size_t ws_size,
                              hipStream_t stream) {
    const float* zr   = (const float*)d_in[0];
    const float* zi   = (const float*)d_in[1];
    const int*   prev = (const int*)d_in[2];
    const float* cb   = (const float*)d_in[3];
    const float* adj  = (const float*)d_in[4];
    float* out = (float*)d_out;

    char* ws = (char*)d_ws;
    size_t off = 0;
    v16bf* Bh16 = (v16bf*)(ws + off); off += (size_t)STILES * KT  * 32 * 32;
    v16i*  B8h  = (v16i*)(ws + off);  off += (size_t)STILES * KT8 * 32 * 64;
    v16i*  B8l  = (v16i*)(ws + off);  off += (size_t)STILES * KT8 * 32 * 64;
    float* cbn  = (float*)(ws + off); off += 64 * 1024;
    v16bf* Az16 = (v16bf*)(ws + off); off += (size_t)MTILES * KT  * 32 * 32;
    v16i*  Az8h = (v16i*)(ws + off);  off += (size_t)MTILES * KT8 * 32 * 64;
    v16i*  Az8l = (v16i*)(ws + off);  off += (size_t)MTILES * KT8 * 32 * 64;

    pack_codebook_kernel<<<STILES, 128, 0, stream>>>(cb, Bh16, B8h, B8l);
    pack_z_kernel<<<MTILES, 128, 0, stream>>>(zr, zi, Az16, Az8h, Az8l);
    cb_norms_kernel<<<NSYM / 4, 128, 0, stream>>>(cb, cbn, out + ZC_FLOATS);
    vq_main_kernel<<<MBLOCKS, 128, 0, stream>>>(zr, zi, prev, cb, adj,
                                                Bh16, B8h, B8l,
                                                Az16, Az8h, Az8l, cbn, out);
}
